// PrismDecoder_75917841924405
// MI455X (gfx1250) — matrix operand order repaired and checked by reading, not compiled, not run on hardware
//
#include <hip/hip_runtime.h>
#include <cstdint>
#include <cstddef>

typedef __attribute__((ext_vector_type(16))) __bf16 v16bf;
typedef __attribute__((ext_vector_type(8)))  float  v8f;

#define N_VERTS 50000
#define N_FACES 100000
#define CDIM    512
#define MTILES  (N_FACES / 16)   // 6250 exact

// ---------------------------------------------------------------------------
// Utility: zero a float buffer
// ---------------------------------------------------------------------------
__global__ void zero_kernel(float* __restrict__ p, int n) {
    int i = blockIdx.x * blockDim.x + threadIdx.x;
    if (i < n) p[i] = 0.0f;
}

// ---------------------------------------------------------------------------
// Gather x[faces], mean over the 3 vertices, convert to bf16.
// One thread per (face, channel). Consecutive threads -> consecutive channels
// of the same face -> coalesced 512-float rows (x is L2-resident: 102 MB).
// ---------------------------------------------------------------------------
__global__ void gather_mean_kernel(const float* __restrict__ x,
                                   const int*   __restrict__ faces,
                                   __bf16*      __restrict__ h0) {
    size_t tid = (size_t)blockIdx.x * blockDim.x + threadIdx.x;
    if (tid >= (size_t)N_FACES * CDIM) return;
    int f = (int)(tid >> 9);
    int c = (int)(tid & (CDIM - 1));
    int i0 = faces[f * 3 + 0];
    int i1 = faces[f * 3 + 1];
    int i2 = faces[f * 3 + 2];
    float v = (x[(size_t)i0 * CDIM + c] +
               x[(size_t)i1 * CDIM + c] +
               x[(size_t)i2 * CDIM + c]) * (1.0f / 3.0f);
    h0[tid] = (__bf16)v;
}

// ---------------------------------------------------------------------------
// Convert f32 weight [K][N] -> bf16 transposed [Npad][K] (zero-pad columns).
// ---------------------------------------------------------------------------
__global__ void convert_wt_kernel(const float* __restrict__ w,
                                  __bf16* __restrict__ wt,
                                  int K, int N, int Npad) {
    int tid = blockIdx.x * blockDim.x + threadIdx.x;
    if (tid >= Npad * K) return;
    int n = tid / K;
    int k = tid - n * K;
    wt[tid] = (n < N) ? (__bf16)w[(size_t)k * N + n] : (__bf16)0.0f;
}

// ---------------------------------------------------------------------------
// WMMA tile loaders (wave32 layouts from CDNA5 ISA §7.12.2).
// A: 16x32 bf16, lane L holds row m=L&15; VGPR v covers K pair
//    k0 + (v>=4?16:0) + (L>>4)*8 + (v&3)*2.
// B: 32x16 bf16 from Wt[N][K] (pre-transposed so K is contiguous); lane L
//    holds column n=L&15; VGPR v covers K pair
//    k0 + (v>=4?16:0) + (L>>4)*8 + (v&3)*2.
// ---------------------------------------------------------------------------
__device__ __forceinline__ v16bf load_tileA(const __bf16* __restrict__ A,
                                            int ldK, int m, int k0, int hi) {
    const __bf16* row = A + (size_t)m * ldK;
    v16bf r;
#pragma unroll
    for (int i = 0; i < 8; ++i) {
        int k = k0 + ((i >> 2) << 4) + hi * 8 + ((i & 3) << 1);
        r[2 * i]     = row[k];
        r[2 * i + 1] = row[k + 1];
    }
    return r;
}

__device__ __forceinline__ v16bf load_tileB(const __bf16* __restrict__ Wt,
                                            int ldK, int n, int k0, int hi) {
    const __bf16* row = Wt + (size_t)n * ldK;
    v16bf r;
#pragma unroll
    for (int i = 0; i < 8; ++i) {
        int k = k0 + ((i >> 2) << 4) + hi * 8 + ((i & 3) << 1);
        r[2 * i]     = row[k];
        r[2 * i + 1] = row[k + 1];
    }
    return r;
}

// ---------------------------------------------------------------------------
// bf16 GEMM + bias + ReLU -> bf16. One wave computes a 16 x (16*NT) strip,
// reusing the A tile NT times per K-step. f32 accumulation via
// v_wmma_f32_16x16x32_bf16.
// ---------------------------------------------------------------------------
template <int NT, bool RELU>
__global__ __launch_bounds__(256) void gemm_bf16_kernel(
    const __bf16* __restrict__ A,    // [M][K] bf16 activations
    const __bf16* __restrict__ Wt,   // [N][K] bf16 transposed weights
    const float*  __restrict__ bias, // [N]
    __bf16*       __restrict__ O,    // [M][N]
    int N, int K) {
    int wid  = blockIdx.x * (blockDim.x >> 5) + (threadIdx.x >> 5);
    int lane = threadIdx.x & 31;
    int strips = N / (16 * NT);
    if (wid >= MTILES * strips) return;   // wave-uniform exit (EXEC stays all-1s)
    int mTile  = wid % MTILES;
    int nStrip = wid / MTILES;
    int m0 = mTile * 16;
    int n0 = nStrip * 16 * NT;
    int mrow = lane & 15;
    int hi   = lane >> 4;

    v8f acc[NT];
#pragma unroll
    for (int t = 0; t < NT; ++t) acc[t] = (v8f){0, 0, 0, 0, 0, 0, 0, 0};

    for (int k0 = 0; k0 < K; k0 += 32) {
        v16bf a = load_tileA(A, K, m0 + mrow, k0, hi);
#pragma unroll
        for (int t = 0; t < NT; ++t) {
            v16bf b = load_tileB(Wt, K, n0 + t * 16 + mrow, k0, hi);
            acc[t] = __builtin_amdgcn_wmma_f32_16x16x32_bf16(
                false, a, false, b, (short)0, acc[t], false, false);
        }
    }

#pragma unroll
    for (int t = 0; t < NT; ++t) {
        int col = n0 + t * 16 + mrow;
        float bv = bias[col];
#pragma unroll
        for (int r = 0; r < 8; ++r) {
            int row = m0 + r + hi * 8;   // C/D layout: VGPR r = rows r / r+8
            float v = acc[t][r] + bv;
            if (RELU) v = v > 0.0f ? v : 0.0f;
            O[(size_t)row * N + col] = (__bf16)v;
        }
    }
}

// ---------------------------------------------------------------------------
// Final layer: K=64, N padded 12->16, no ReLU, f32 output [M][12].
// ---------------------------------------------------------------------------
__global__ __launch_bounds__(256) void gemm_last_kernel(
    const __bf16* __restrict__ A,    // [M][64]
    const __bf16* __restrict__ Wt,   // [16][64] (cols 12..15 zero)
    const float*  __restrict__ bias, // [12]
    float*        __restrict__ O) {  // [M][12]
    const int K = 64;
    int wid  = blockIdx.x * (blockDim.x >> 5) + (threadIdx.x >> 5);
    int lane = threadIdx.x & 31;
    if (wid >= MTILES) return;
    int m0 = wid * 16;
    int mrow = lane & 15;
    int hi   = lane >> 4;

    v8f acc = (v8f){0, 0, 0, 0, 0, 0, 0, 0};
#pragma unroll
    for (int k0 = 0; k0 < K; k0 += 32) {
        v16bf a = load_tileA(A, K, m0 + mrow, k0, hi);
        v16bf b = load_tileB(Wt, K, mrow, k0, hi);
        acc = __builtin_amdgcn_wmma_f32_16x16x32_bf16(
            false, a, false, b, (short)0, acc, false, false);
    }

    int col = mrow;
    if (col < 12) {
        float bv = bias[col];
#pragma unroll
        for (int r = 0; r < 8; ++r) {
            int row = m0 + r + hi * 8;
            O[(size_t)row * 12 + col] = acc[r] + bv;
        }
    }
}

// ---------------------------------------------------------------------------
// Per-face special Procrustes (3x3), prism transform, outputs + segment sums.
// S = M^T M -> Jacobi eigen (6 sweeps) -> V, sigma; U = normalize(M v_i);
// det fix on the last (smallest-sigma) column, matching fix=[1,1,d].
// ---------------------------------------------------------------------------
__global__ void procrustes_kernel(const float* __restrict__ h6,    // [F][12]
                                  const float* __restrict__ verts, // [V][3]
                                  const int*   __restrict__ faces, // [F][3]
                                  float* __restrict__ prism_out,   // [F][3][3]
                                  float* __restrict__ rot_out,     // [F][3][3]
                                  float* __restrict__ sums,        // [V][3]
                                  float* __restrict__ cnt) {       // [V]
    int f = blockIdx.x * blockDim.x + threadIdx.x;
    if (f >= N_FACES) return;
    const float* h = h6 + (size_t)f * 12;

    float M[3][3];
#pragma unroll
    for (int j = 0; j < 3; ++j)
#pragma unroll
        for (int k = 0; k < 3; ++k) M[j][k] = h[3 * j + k];

    // S = M^T M (symmetric)
    float S[3][3];
#pragma unroll
    for (int a = 0; a < 3; ++a)
#pragma unroll
        for (int b = 0; b < 3; ++b)
            S[a][b] = M[0][a] * M[0][b] + M[1][a] * M[1][b] + M[2][a] * M[2][b];

    float V[3][3] = {{1, 0, 0}, {0, 1, 0}, {0, 0, 1}};

#pragma unroll
    for (int sweep = 0; sweep < 6; ++sweep) {
#pragma unroll
        for (int pair = 0; pair < 3; ++pair) {
            int p = (pair == 2) ? 1 : 0;
            int q = (pair == 0) ? 1 : 2;
            float apq = S[p][q];
            if (fabsf(apq) > 1e-20f) {
                float app = S[p][p], aqq = S[q][q];
                float tau = (aqq - app) / (2.0f * apq);
                float t   = copysignf(1.0f, tau) /
                            (fabsf(tau) + sqrtf(1.0f + tau * tau));
                float c = rsqrtf(1.0f + t * t);
                float s = t * c;
                int r = 3 - p - q;
                float srp = S[r][p], srq = S[r][q];
                S[p][p] = app - t * apq;
                S[q][q] = aqq + t * apq;
                S[p][q] = 0.0f; S[q][p] = 0.0f;
                S[r][p] = c * srp - s * srq; S[p][r] = S[r][p];
                S[r][q] = s * srp + c * srq; S[q][r] = S[r][q];
#pragma unroll
                for (int i = 0; i < 3; ++i) {
                    float vip = V[i][p], viq = V[i][q];
                    V[i][p] = c * vip - s * viq;
                    V[i][q] = s * vip + c * viq;
                }
            }
        }
    }

    // Sort eigenpairs descending (columns of V follow).
    float e[3] = {S[0][0], S[1][1], S[2][2]};
#pragma unroll
    for (int pass = 0; pass < 3; ++pass) {
        int i = (pass == 1) ? 1 : 0;
        int j = (pass == 0) ? 1 : 2;
        if (e[i] < e[j]) {
            float te = e[i]; e[i] = e[j]; e[j] = te;
#pragma unroll
            for (int r = 0; r < 3; ++r) {
                float tv = V[r][i]; V[r][i] = V[r][j]; V[r][j] = tv;
            }
        }
    }

    // U columns: normalize(M v_i); cross-product fallback for the last.
    float U[3][3];
#pragma unroll
    for (int i = 0; i < 3; ++i) {
        float u0 = M[0][0] * V[0][i] + M[0][1] * V[1][i] + M[0][2] * V[2][i];
        float u1 = M[1][0] * V[0][i] + M[1][1] * V[1][i] + M[1][2] * V[2][i];
        float u2 = M[2][0] * V[0][i] + M[2][1] * V[1][i] + M[2][2] * V[2][i];
        float len2 = u0 * u0 + u1 * u1 + u2 * u2;
        if (len2 > 1e-24f) {
            float inv = rsqrtf(len2);
            U[0][i] = u0 * inv; U[1][i] = u1 * inv; U[2][i] = u2 * inv;
        } else if (i == 2) {
            U[0][2] = U[1][0] * U[2][1] - U[2][0] * U[1][1];
            U[1][2] = U[2][0] * U[0][1] - U[0][0] * U[2][1];
            U[2][2] = U[0][0] * U[1][1] - U[1][0] * U[0][1];
        } else {
            U[0][i] = (i == 0) ? 1.0f : 0.0f;
            U[1][i] = (i == 1) ? 1.0f : 0.0f;
            U[2][i] = 0.0f;
        }
    }

    // R0 = U V^T ; det sign fix on last column of U.
    float R[3][3];
#pragma unroll
    for (int j = 0; j < 3; ++j)
#pragma unroll
        for (int k = 0; k < 3; ++k)
            R[j][k] = U[j][0] * V[k][0] + U[j][1] * V[k][1] + U[j][2] * V[k][2];

    float det = R[0][0] * (R[1][1] * R[2][2] - R[1][2] * R[2][1])
              - R[0][1] * (R[1][0] * R[2][2] - R[1][2] * R[2][0])
              + R[0][2] * (R[1][0] * R[2][1] - R[1][1] * R[2][0]);
    float fixv = (det < 0.0f) ? -2.0f : 0.0f;
#pragma unroll
    for (int j = 0; j < 3; ++j)
#pragma unroll
        for (int k = 0; k < 3; ++k)
            R[j][k] += fixv * U[j][2] * V[k][2];

#pragma unroll
    for (int j = 0; j < 3; ++j)
#pragma unroll
        for (int k = 0; k < 3; ++k)
            rot_out[(size_t)f * 9 + j * 3 + k] = R[j][k];

    float t0 = h[9], t1 = h[10], t2 = h[11];
#pragma unroll
    for (int v = 0; v < 3; ++v) {
        int vi = faces[f * 3 + v];
        float b0 = verts[(size_t)vi * 3 + 0];
        float b1 = verts[(size_t)vi * 3 + 1];
        float b2 = verts[(size_t)vi * 3 + 2];
        float p0 = b0 * R[0][0] + b1 * R[1][0] + b2 * R[2][0] + t0;
        float p1 = b0 * R[0][1] + b1 * R[1][1] + b2 * R[2][1] + t1;
        float p2 = b0 * R[0][2] + b1 * R[1][2] + b2 * R[2][2] + t2;
        prism_out[(size_t)f * 9 + v * 3 + 0] = p0;
        prism_out[(size_t)f * 9 + v * 3 + 1] = p1;
        prism_out[(size_t)f * 9 + v * 3 + 2] = p2;
        atomicAdd(&sums[(size_t)vi * 3 + 0], p0);
        atomicAdd(&sums[(size_t)vi * 3 + 1], p1);
        atomicAdd(&sums[(size_t)vi * 3 + 2], p2);
        atomicAdd(&cnt[vi], 1.0f);
    }
}

__global__ void finalize_kernel(const float* __restrict__ sums,
                                const float* __restrict__ cnt,
                                float* __restrict__ feat) {
    int i = blockIdx.x * blockDim.x + threadIdx.x;
    if (i >= N_VERTS) return;
    float c = cnt[i];
    c = c < 1.0f ? 1.0f : c;
    float inv = 1.0f / c;
    feat[(size_t)i * 3 + 0] = sums[(size_t)i * 3 + 0] * inv;
    feat[(size_t)i * 3 + 1] = sums[(size_t)i * 3 + 1] * inv;
    feat[(size_t)i * 3 + 2] = sums[(size_t)i * 3 + 2] * inv;
}

// ---------------------------------------------------------------------------
// Host launcher
// ---------------------------------------------------------------------------
extern "C" void kernel_launch(void* const* d_in, const int* in_sizes, int n_in,
                              void* d_out, int out_size, void* d_ws, size_t ws_size,
                              hipStream_t stream) {
    const float* x     = (const float*)d_in[0];
    const float* verts = (const float*)d_in[1];
    const int*   faces = (const int*)d_in[2];
    const float* w[6]  = {(const float*)d_in[3],  (const float*)d_in[5],
                          (const float*)d_in[7],  (const float*)d_in[9],
                          (const float*)d_in[11], (const float*)d_in[13]};
    const float* b[6]  = {(const float*)d_in[4],  (const float*)d_in[6],
                          (const float*)d_in[8],  (const float*)d_in[10],
                          (const float*)d_in[12], (const float*)d_in[14]};

    // Workspace carve-up (256-byte aligned)
    char* ws = (char*)d_ws;
    size_t off = 0;
    auto carve = [&](size_t bytes) -> void* {
        void* p = ws + off;
        off += (bytes + 255) & ~(size_t)255;
        return p;
    };
    __bf16* hA   = (__bf16*)carve((size_t)N_FACES * CDIM * 2);
    __bf16* hB   = (__bf16*)carve((size_t)N_FACES * CDIM * 2);
    __bf16* wt1  = (__bf16*)carve((size_t)512 * 512 * 2);
    __bf16* wt2  = (__bf16*)carve((size_t)512 * 512 * 2);
    __bf16* wt3  = (__bf16*)carve((size_t)256 * 512 * 2);
    __bf16* wt4  = (__bf16*)carve((size_t)128 * 256 * 2);
    __bf16* wt5  = (__bf16*)carve((size_t)64 * 128 * 2);
    __bf16* wt6  = (__bf16*)carve((size_t)16 * 64 * 2);
    float*  h6   = (float*)carve((size_t)N_FACES * 12 * 4);
    float*  sums = (float*)carve((size_t)N_VERTS * 3 * 4);
    float*  cnt  = (float*)carve((size_t)N_VERTS * 4);

    float* out_feat  = (float*)d_out;                       // (1, V, 3)
    float* out_prism = out_feat + (size_t)N_VERTS * 3;      // (F, 3, 3)
    float* out_rot   = out_prism + (size_t)N_FACES * 9;     // (F, 3, 3)

    // Zero segment accumulators
    zero_kernel<<<(N_VERTS * 3 + 255) / 256, 256, 0, stream>>>(sums, N_VERTS * 3);
    zero_kernel<<<(N_VERTS + 255) / 256, 256, 0, stream>>>(cnt, N_VERTS);

    // Gather + mean -> bf16
    {
        size_t total = (size_t)N_FACES * CDIM;
        gather_mean_kernel<<<(unsigned)((total + 255) / 256), 256, 0, stream>>>(x, faces, hA);
    }

    // Transpose + convert weights to bf16 [N][K]
    convert_wt_kernel<<<(512 * 512 + 255) / 256, 256, 0, stream>>>(w[0], wt1, 512, 512, 512);
    convert_wt_kernel<<<(512 * 512 + 255) / 256, 256, 0, stream>>>(w[1], wt2, 512, 512, 512);
    convert_wt_kernel<<<(256 * 512 + 255) / 256, 256, 0, stream>>>(w[2], wt3, 512, 256, 256);
    convert_wt_kernel<<<(128 * 256 + 255) / 256, 256, 0, stream>>>(w[3], wt4, 256, 128, 128);
    convert_wt_kernel<<<(64 * 128 + 255) / 256, 256, 0, stream>>>(w[4], wt5, 128, 64, 64);
    convert_wt_kernel<<<(16 * 64 + 255) / 256, 256, 0, stream>>>(w[5], wt6, 64, 12, 16);

    // MLP: 8 waves/block; total waves = MTILES * (N/64)
    gemm_bf16_kernel<4, true><<<(MTILES * 8 + 7) / 8, 256, 0, stream>>>(hA, wt1, b[0], hB, 512, 512);
    gemm_bf16_kernel<4, true><<<(MTILES * 8 + 7) / 8, 256, 0, stream>>>(hB, wt2, b[1], hA, 512, 512);
    gemm_bf16_kernel<4, true><<<(MTILES * 4 + 7) / 8, 256, 0, stream>>>(hA, wt3, b[2], hB, 256, 512);
    gemm_bf16_kernel<4, true><<<(MTILES * 2 + 7) / 8, 256, 0, stream>>>(hB, wt4, b[3], hA, 128, 256);
    gemm_bf16_kernel<4, true><<<(MTILES * 1 + 7) / 8, 256, 0, stream>>>(hA, wt5, b[4], hB, 64, 128);
    gemm_last_kernel<<<(MTILES + 7) / 8, 256, 0, stream>>>(hB, wt6, b[5], h6);

    // Procrustes + prism + scatter
    procrustes_kernel<<<(N_FACES + 255) / 256, 256, 0, stream>>>(
        h6, verts, faces, out_prism, out_rot, sums, cnt);

    // Vertex feature average
    finalize_kernel<<<(N_VERTS + 255) / 256, 256, 0, stream>>>(sums, cnt, out_feat);

    (void)in_sizes; (void)n_in; (void)out_size; (void)ws_size;
}
